// RoPEAttention_27865747816591
// MI455X (gfx1250) — matrix operand order repaired and checked by reading, not compiled, not run on hardware
//
#include <hip/hip_runtime.h>

// ---------------------------------------------------------------------------
// RoPE multi-head attention, all GEMMs via v_wmma_f32_16x16x32_bf16 (CDNA5).
// B=2, N=2048, D=1024, H=16, HD=64.
// All GEMMs take B column-major (BT = [N][K] row-major) so LDS staging is
// pure b128 copies — no in-kernel transpose packing.
// ---------------------------------------------------------------------------

typedef __attribute__((ext_vector_type(16))) __bf16 v16bf;
typedef __attribute__((ext_vector_type(8)))  float  v8f;
typedef __attribute__((ext_vector_type(8)))  unsigned int v8u;

__device__ __forceinline__ unsigned short f2bf(float f) {
    unsigned u = __builtin_bit_cast(unsigned, f);
    unsigned r = u + 0x7FFFu + ((u >> 16) & 1u);   // round-to-nearest-even
    return (unsigned short)(r >> 16);
}
__device__ __forceinline__ float bf2f(unsigned short h) {
    return __builtin_bit_cast(float, (unsigned)h << 16);
}

// ---------------------------------------------------------------------------
// Elementwise f32 -> bf16 cast
// ---------------------------------------------------------------------------
__global__ __launch_bounds__(256)
void cast_to_bf16(const float* __restrict__ src, unsigned short* __restrict__ dst, int n) {
    int i = blockIdx.x * 256 + threadIdx.x;
    if (i < n) dst[i] = f2bf(src[i]);
}

// ---------------------------------------------------------------------------
// Tiled transpose + cast: src f32 [R][C] row-major -> dst bf16 [C][R].
// 32x32 tiles via LDS; coalesced reads and writes. R%32==0, C%32==0.
// ---------------------------------------------------------------------------
__global__ __launch_bounds__(256)
void cast_transpose_bf16(const float* __restrict__ src, unsigned short* __restrict__ dst,
                         int R, int C)
{
    __shared__ unsigned short t[32][33];
    const int tx = threadIdx.x & 31;
    const int ty = threadIdx.x >> 5;         // 0..7
    const int r0 = blockIdx.y * 32;
    const int c0 = blockIdx.x * 32;
    #pragma unroll
    for (int i = 0; i < 4; ++i)
        t[ty + i * 8][tx] = f2bf(src[(size_t)(r0 + ty + i * 8) * C + c0 + tx]);
    __syncthreads();
    #pragma unroll
    for (int i = 0; i < 4; ++i)
        dst[(size_t)(c0 + ty + i * 8) * R + r0 + tx] = t[tx][ty + i * 8];
}

// ---------------------------------------------------------------------------
// Tiled bf16 GEMM, f32 accumulate via WMMA.
//   C[M,N] = alpha * (A[M,K] @ B[K,N]) + bias[N],  B given as BT[N][K].
//
// Block: 256 threads = 8 wave32 waves. Block tile 64 x BN, K-step 32.
// Wave grid: (64/MT) rows x (BN/32) cols; wave tile MT x 32.
//   BN=128, MT=32: 2x4 waves, 4 WMMA per K-step per wave.
//   BN=64,  MT=16: 4x2 waves, 2 WMMA per K-step (used for N=64 PV GEMM).
// LDS: A tile [64][40] halves, BT tile [BN][40] halves; stride 40 halves
// (80 B) keeps b128 rows 16B-aligned and fragment reads conflict-free.
// Requires M%64==0, N%BN==0, K%32==0 (true for all uses here).
// ---------------------------------------------------------------------------
#define GBM 64
#define GBK 32
#define LDS_STRIDE 40

template <bool OUT_BF16, int BN, int MT>
__global__ __launch_bounds__(256)
void gemm_bf16(const unsigned short* __restrict__ A, int lda,
               const unsigned short* __restrict__ BT, int ldbt,
               float* __restrict__ Cf, unsigned short* __restrict__ Cb, int ldc,
               const float* __restrict__ bias,
               int M, int N, int K, float alpha)
{
    constexpr int AF = MT / 16;        // A fragments per wave
    constexpr int WN = BN / 32;        // waves along N
    static_assert((GBM / MT) * WN == 8, "wave grid must be 8 waves");

    __shared__ unsigned short lA[GBM * LDS_STRIDE];
    __shared__ unsigned short lB[BN * LDS_STRIDE];   // [n][k]

    const int tid  = threadIdx.x;
    const int wave = tid >> 5;
    const int lane = tid & 31;
    const int wr   = wave / WN;        // wave row in block tile
    const int wc   = wave % WN;        // wave col in block tile
    const int m0   = blockIdx.y * GBM;
    const int n0   = blockIdx.x * BN;

    // tile loaders: one 16B (8 halves) per thread per 64 rows
    const int lr = tid >> 2;                // 0..63 row
    const int lc = (tid & 3) << 3;          // 0,8,16,24 (halves)

    const int m  = lane & 15;               // frag row (A) / col (B,C)
    const int kh = lane >> 4;               // half-wave selector

    v8f acc[AF][2] = {};

    for (int k0 = 0; k0 < K; k0 += GBK) {
        // ---- stage A tile (b128 copies) ----
        uint4 a4 = *reinterpret_cast<const uint4*>(
            &A[(size_t)(m0 + lr) * lda + k0 + lc]);
        *reinterpret_cast<uint4*>(&lA[lr * LDS_STRIDE + lc]) = a4;

        // ---- stage BT tile (b128 copies) ----
        #pragma unroll
        for (int i = 0; i < BN / 64; ++i) {
            int r = lr + 64 * i;
            uint4 b4 = *reinterpret_cast<const uint4*>(
                &BT[(size_t)(n0 + r) * ldbt + k0 + lc]);
            *reinterpret_cast<uint4*>(&lB[r * LDS_STRIDE + lc]) = b4;
        }

        if (k0 + GBK < K) {   // prefetch next K-tile -> global_prefetch_b8
            __builtin_prefetch(&A[(size_t)(m0 + lr) * lda + k0 + GBK + lc], 0, 1);
            __builtin_prefetch(&BT[(size_t)(n0 + lr) * ldbt + k0 + GBK + lc], 0, 1);
        }
        __syncthreads();

        // ---- build fragments per ISA 16-bit layouts ----
        v16bf afrag[AF];
        #pragma unroll
        for (int a = 0; a < AF; ++a) {
            const unsigned short* arow = &lA[(wr * MT + a * 16 + m) * LDS_STRIDE];
            v8u u;
            #pragma unroll
            for (int v = 0; v < 8; ++v) {
                // A 16x32: k = (v>=4)*16 + kh*8 + (v%4)*2
                int ka = ((v & 4) << 2) + (kh << 3) + ((v & 3) << 1);
                u[v] = *reinterpret_cast<const unsigned int*>(&arow[ka]);
            }
            afrag[a] = __builtin_bit_cast(v16bf, u);
        }
        v16bf bfrag[2];
        #pragma unroll
        for (int t = 0; t < 2; ++t) {
            const unsigned short* brow = &lB[(wc * 32 + t * 16 + m) * LDS_STRIDE];
            v8u u;
            #pragma unroll
            for (int v = 0; v < 8; ++v) {
                // B 32x16: k = (v>=4)*16 + kh*8 + (v%4)*2
                u[v] = *reinterpret_cast<const unsigned int*>(&brow[((v & 4) << 2) + (kh << 3) + ((v & 3) << 1)]);
            }
            bfrag[t] = __builtin_bit_cast(v16bf, u);
        }

        #pragma unroll
        for (int a = 0; a < AF; ++a)
            #pragma unroll
            for (int t = 0; t < 2; ++t)
                acc[a][t] = __builtin_amdgcn_wmma_f32_16x16x32_bf16(
                    false, afrag[a], false, bfrag[t], (short)0, acc[a][t],
                    false, false);

        __syncthreads();
    }

    // ---- epilogue: C/D layout — VGPR r holds row r + 8*kh, col = lane&15 ----
    #pragma unroll
    for (int a = 0; a < AF; ++a) {
        #pragma unroll
        for (int t = 0; t < 2; ++t) {
            int cc = n0 + wc * 32 + t * 16 + m;
            float bv = bias ? bias[cc] : 0.0f;
            #pragma unroll
            for (int r = 0; r < 8; ++r) {
                int row = m0 + wr * MT + a * 16 + (kh << 3) + r;
                float v = alpha * acc[a][t][r] + bv;
                if (OUT_BF16) Cb[(size_t)row * ldc + cc] = f2bf(v);
                else          Cf[(size_t)row * ldc + cc] = v;
            }
        }
    }
}

// ---------------------------------------------------------------------------
// RoPE + head split. qkv: bf16 [b*n, 3*1024]. One thread per (b,n,h,j<32).
// Writes q,k in [b,h,n,64] (k's col-major form for QK^T GEMM is its natural
// layout) and v TRANSPOSED in [b,h,64,n] (col-major form for PV GEMM).
// ---------------------------------------------------------------------------
__global__ __launch_bounds__(256)
void rope_split(const unsigned short* __restrict__ qkv,
                unsigned short* __restrict__ qb,
                unsigned short* __restrict__ kb,
                unsigned short* __restrict__ vtb)
{
    int idx = blockIdx.x * 256 + threadIdx.x;    // < 2*2048*16*32
    int j = idx & 31;
    int h = (idx >> 5) & 15;
    int n = (idx >> 9) & 2047;
    int b = idx >> 20;
    const int N = 2048, D = 1024, HD = 64;

    size_t row = (size_t)(b * N + n) * (3 * D);
    int cb = h * HD + j;
    float q1 = bf2f(qkv[row + cb]);
    float q2 = bf2f(qkv[row + cb + 32]);
    float k1 = bf2f(qkv[row + D + cb]);
    float k2 = bf2f(qkv[row + D + cb + 32]);
    unsigned short v1 = qkv[row + 2 * D + cb];
    unsigned short v2 = qkv[row + 2 * D + cb + 32];

    // theta_j = 10000^(-j/32) = exp(-j * ln(10000)/32)
    float theta = __expf(-0.28782313662425572f * (float)j);
    float ang = theta * (float)n;
    float s, c;
    __sincosf(ang, &s, &c);

    size_t qoff = ((size_t)(b * 16 + h) * N + n) * HD + j;
    qb[qoff]      = f2bf(c * q1 - s * q2);
    qb[qoff + 32] = f2bf(s * q1 + c * q2);
    kb[qoff]      = f2bf(c * k1 - s * k2);
    kb[qoff + 32] = f2bf(s * k1 + c * k2);

    size_t voff = ((size_t)(b * 16 + h) * HD + j) * N + n;
    vtb[voff]                  = v1;
    vtb[voff + 32 * (size_t)N] = v2;
}

// ---------------------------------------------------------------------------
// Row softmax: f32 scores [rows, ncols] -> bf16 probs. One block per row.
// ---------------------------------------------------------------------------
__global__ __launch_bounds__(256)
void softmax_rows(const float* __restrict__ S, unsigned short* __restrict__ P, int ncols)
{
    __shared__ float red[256];
    const int row = blockIdx.x;
    const int tid = threadIdx.x;
    const float* s = S + (size_t)row * ncols;

    float mx = -3.4e38f;
    for (int c = tid; c < ncols; c += 256) mx = fmaxf(mx, s[c]);
    red[tid] = mx; __syncthreads();
    for (int off = 128; off > 0; off >>= 1) {
        if (tid < off) red[tid] = fmaxf(red[tid], red[tid + off]);
        __syncthreads();
    }
    mx = red[0]; __syncthreads();

    float e[8];
    float sum = 0.0f;
    int i = 0;
    for (int c = tid; c < ncols; c += 256, ++i) {
        float v = __expf(s[c] - mx);
        e[i] = v; sum += v;
    }
    red[tid] = sum; __syncthreads();
    for (int off = 128; off > 0; off >>= 1) {
        if (tid < off) red[tid] += red[tid + off];
        __syncthreads();
    }
    float inv = 1.0f / red[0];
    i = 0;
    for (int c = tid; c < ncols; c += 256, ++i)
        P[(size_t)row * ncols + c] = f2bf(e[i] * inv);
}

// ---------------------------------------------------------------------------
// Host-side orchestration
// ---------------------------------------------------------------------------
extern "C" void kernel_launch(void* const* d_in, const int* in_sizes, int n_in,
                              void* d_out, int out_size, void* d_ws, size_t ws_size,
                              hipStream_t stream)
{
    const int B = 2, N = 2048, D = 1024, H = 16, HD = 64;
    const int BN_TOK = B * N;          // 4096 token rows

    const float* x    = (const float*)d_in[0];
    const float* Wqkv = (const float*)d_in[1];
    const float* bqkv = (const float*)d_in[2];
    const float* Wout = (const float*)d_in[3];
    const float* bout = (const float*)d_in[4];
    float* out = (float*)d_out;

    // workspace layout (bytes)
    char* ws = (char*)d_ws;
    size_t off = 0;
    unsigned short* xb     = (unsigned short*)(ws + off); off += (size_t)BN_TOK * D * 2;       // 8 MB
    unsigned short* wqkvt  = (unsigned short*)(ws + off); off += (size_t)D * 3 * D * 2;        // 6 MB (transposed)
    unsigned short* woutt  = (unsigned short*)(ws + off); off += (size_t)D * D * 2;            // 2 MB (transposed)
    unsigned short* qkvb   = (unsigned short*)(ws + off); off += (size_t)BN_TOK * 3 * D * 2;   // 24 MB
    unsigned short* qb     = (unsigned short*)(ws + off); off += (size_t)B * H * N * HD * 2;   // 8 MB
    unsigned short* kb     = (unsigned short*)(ws + off); off += (size_t)B * H * N * HD * 2;   // 8 MB
    unsigned short* vtb    = (unsigned short*)(ws + off); off += (size_t)B * H * HD * N * 2;   // 8 MB
    unsigned short* ctxb   = (unsigned short*)(ws + off); off += (size_t)BN_TOK * D * 2;       // 8 MB
    float*          sc     = (float*)(ws + off);          off += (size_t)N * N * 4;            // 16 MB
    unsigned short* pb     = (unsigned short*)(ws + off); off += (size_t)N * N * 2;            // 8 MB
    if (ws_size < off) return;   // ~96 MB required

    // 1) casts to bf16 (weights transposed to column-major)
    cast_to_bf16<<<(BN_TOK * D + 255) / 256, 256, 0, stream>>>(x, xb, BN_TOK * D);
    cast_transpose_bf16<<<dim3(3 * D / 32, D / 32), 256, 0, stream>>>(Wqkv, wqkvt, D, 3 * D);
    cast_transpose_bf16<<<dim3(D / 32, D / 32), 256, 0, stream>>>(Wout, woutt, D, D);

    // 2) QKV projection: [4096,1024] @ [1024,3072] + b_qkv -> bf16
    gemm_bf16<true, 128, 32><<<dim3(3 * D / 128, BN_TOK / GBM), 256, 0, stream>>>(
        xb, D, wqkvt, D, nullptr, qkvb, 3 * D, bqkv, BN_TOK, 3 * D, D, 1.0f);

    // 3) RoPE + split into q [b,h,n,hd], k [b,h,n,hd], v^T [b,h,hd,n]
    rope_split<<<(B * N * H * 32) / 256, 256, 0, stream>>>(qkvb, qb, kb, vtb);

    // 4) per-(b,h) attention; stream order lets sc/pb buffers be reused
    const float inv_sqrt_hd = 0.125f;   // 1/sqrt(64)
    for (int bh = 0; bh < B * H; ++bh) {
        const unsigned short* qh  = qb  + (size_t)bh * N * HD;
        const unsigned short* kh_ = kb  + (size_t)bh * N * HD;
        const unsigned short* vth = vtb + (size_t)bh * HD * N;

        // scores = q @ k^T * inv_sqrt_hd : BT = k itself ([n][64])
        gemm_bf16<false, 128, 32><<<dim3(N / 128, N / GBM), 256, 0, stream>>>(
            qh, HD, kh_, HD, sc, nullptr, N, nullptr, N, N, HD, inv_sqrt_hd);

        // softmax rows -> bf16 probs
        softmax_rows<<<N, 256, 0, stream>>>(sc, pb, N);

        // ctx = probs @ v : BT = v^T ([64 n][2048 k]); output scattered into
        // [b, n, h*64 .. h*64+63] of the [b*n, 1024] context matrix
        int b = bh >> 4, h = bh & 15;
        unsigned short* ctxp = ctxb + (size_t)b * N * D + (size_t)h * HD;
        gemm_bf16<true, 64, 16><<<dim3(HD / 64, N / GBM), 256, 0, stream>>>(
            pb, N, vth, N, nullptr, ctxp, D, nullptr, N, HD, N, 1.0f);
    }

    // 5) output projection: [4096,1024] @ [1024,1024] + b_out -> f32 d_out
    gemm_bf16<false, 128, 32><<<dim3(D / 128, BN_TOK / GBM), 256, 0, stream>>>(
        ctxb, D, woutt, D, out, nullptr, D, bout, BN_TOK, D, D, 1.0f);
}
